// ReprogrammingLayer_77567109366446
// MI455X (gfx1250) — matrix operand order repaired and checked by reading, not compiled, not run on hardware
//
#include <hip/hip_runtime.h>

#define DI static __device__ __forceinline__

typedef __attribute__((ext_vector_type(8)))  float  v8f;
typedef __attribute__((ext_vector_type(4)))  float  f32x4;
typedef __attribute__((ext_vector_type(16))) __bf16 v16bf;
typedef __attribute__((ext_vector_type(8)))  __bf16 bf16x8;
typedef __attribute__((ext_vector_type(4)))  __bf16 bf16x4;
typedef __attribute__((ext_vector_type(4)))  int    i32x4;

// Problem constants
constexpr int Bsz  = 16;
constexpr int Lseq = 512;
constexpr int DM   = 768;
constexpr int Sseq = 2048;
constexpr int DLL  = 4096;
constexpr int NH   = 8;
constexpr int EH   = 96;

// ---------------------------------------------------------------------------
// CDNA5 async global->LDS staging (ASYNCcnt path), with safe fallback.
// Builtin signature (probed from clang diagnostic):
//   void __builtin_amdgcn_global_load_async_to_lds_b128(
//       int4 addrspace(1)* gsrc, int4 addrspace(3)* ldst, imm offset, imm cpol)
// ---------------------------------------------------------------------------
#if defined(__HIP_DEVICE_COMPILE__) &&                                         \
    __has_builtin(__builtin_amdgcn_global_load_async_to_lds_b128) &&           \
    __has_builtin(__builtin_amdgcn_s_wait_asynccnt)
#define USE_ASYNC 1
#else
#define USE_ASYNC 0
#endif

#if USE_ASYNC
typedef __attribute__((address_space(1))) i32x4 as1_i32x4;
typedef __attribute__((address_space(3))) i32x4 as3_i32x4;
DI void g2lds16(const __bf16* g, __bf16* l) {
  // 16 bytes, per-lane addresses; lands directly in LDS, tracked by ASYNCcnt.
  __builtin_amdgcn_global_load_async_to_lds_b128(
      (as1_i32x4*)(unsigned long long)g,
      (as3_i32x4*)(unsigned)(unsigned long long)l, 0, 0);
}
#define WAIT_ASYNC(n) __builtin_amdgcn_s_wait_asynccnt(n)
#else
DI void g2lds16(const __bf16* g, __bf16* l) {
  *(bf16x8*)l = *(const bf16x8*)g;
}
#define WAIT_ASYNC(n) do { } while (0)
#endif

DI v8f wmma_bf16(v16bf a, v16bf b, v8f c) {
  return __builtin_amdgcn_wmma_f32_16x16x32_bf16(false, a, false, b, (short)0, c,
                                                 false, false);
}

// B-operand fragment: gapped layout, halfs [0..7] and [16..23] rel. to p
DI v16bf ldB(const __bf16* p) {
  bf16x8 a = *(const bf16x8*)p;
  bf16x8 b = *(const bf16x8*)(p + 16);
  v16bf r;
#pragma unroll
  for (int i = 0; i < 8; ++i) { r[i] = a[i]; r[i + 8] = b[i]; }
  return r;
}

// A-operand fragment: gapped layout, halfs [0..7] and [16..23] rel. to p
DI v16bf ldA(const __bf16* p) {
  bf16x8 a = *(const bf16x8*)p;
  bf16x8 b = *(const bf16x8*)(p + 16);
  v16bf r;
#pragma unroll
  for (int i = 0; i < 8; ++i) { r[i] = a[i]; r[i + 8] = b[i]; }
  return r;
}

// ---------------------------------------------------------------------------
// Weight transpose + fp32 -> bf16 convert:  W (R x C) -> WT (C x R) bf16
// ---------------------------------------------------------------------------
__global__ void wtrans_kernel(const float* __restrict__ W, __bf16* __restrict__ WT,
                              int R, int C) {
  __shared__ float t[32][33];
  const int tx = threadIdx.x, ty = threadIdx.y;
  const int c  = blockIdx.x * 32 + tx;
  const int r0 = blockIdx.y * 32;
  for (int i = ty; i < 32; i += 8) t[i][tx] = W[(size_t)(r0 + i) * C + c];
  __syncthreads();
  const int r  = r0 + tx;
  const int c0 = blockIdx.x * 32;
  for (int i = ty; i < 32; i += 8)
    WT[(size_t)(c0 + i) * R + r] = (__bf16)t[tx][i];
}

// ---------------------------------------------------------------------------
// Tiled WMMA GEMM:  Out[M][N] = A[M][K] * Bt[N][K]^T + bias[N]
//   AT = float (convert on load, register-prefetch double buffer) or
//        __bf16 (async global->LDS double buffer); OT = __bf16 or float
//   128x128 block tile, BK=32, 8 waves (4x2), wave tile 32x64
//   LDS tiles are ping-pong double-buffered; next tile's async fills overlap
//   the current tile's WMMAs, fenced with s_wait_asynccnt.
// ---------------------------------------------------------------------------
template <typename AT, typename OT>
__global__ __launch_bounds__(256, 1)
void gemm_wmma_kernel(const AT* __restrict__ A, const __bf16* __restrict__ Bt,
                      const float* __restrict__ bias, OT* __restrict__ Out,
                      int M, int N, int K) {
  constexpr int STR  = 48;                // LDS row stride (halfs), 16B multiple
  constexpr int TILE = 128 * STR;         // halfs per tile
  __shared__ __bf16 As[2 * TILE];
  __shared__ __bf16 Bs[2 * TILE];
  const int tid  = threadIdx.x;
  const int lane = tid & 31;
  const int w    = tid >> 5;
  const int wm   = w >> 1;                // 0..3  (32 rows each)
  const int wn   = w & 1;                 // 0..1  (64 cols each)
  const int hl   = lane & 15;
  const int hi   = lane >> 4;             // half-wave select
  const int m0   = blockIdx.y * 128;
  const int n0   = blockIdx.x * 128;
  const int nk   = K / 32;

  // staging thread mapping
  const int ar  = tid >> 3;               // float-A path: rows 0..31 (+32*i)
  const int ac4 = (tid & 7) * 4;
  const int br  = tid >> 1;               // bf16 rows 0..127
  const int bq  = (tid & 1) * 16;

  v8f acc[2][4];
#pragma unroll
  for (int mi = 0; mi < 2; ++mi)
#pragma unroll
    for (int ni = 0; ni < 4; ++ni) acc[mi][ni] = {};

  // ---- prologue: stage k-step 0 ----
  f32x4 areg[4];
  if constexpr (sizeof(AT) == 4) {
#pragma unroll
    for (int i = 0; i < 4; ++i)
      areg[i] = *(const f32x4*)((const float*)A + (size_t)(m0 + ar + 32 * i) * K + ac4);
  } else {
    const __bf16* src = (const __bf16*)A + (size_t)(m0 + br) * K + bq;
    g2lds16(src,     As + br * STR + bq);
    g2lds16(src + 8, As + br * STR + bq + 8);
  }
  {
    const __bf16* src = Bt + (size_t)(n0 + br) * K + bq;
    g2lds16(src,     Bs + br * STR + bq);
    g2lds16(src + 8, Bs + br * STR + bq + 8);
  }

  for (int ik = 0; ik < nk; ++ik) {
    const int cur = ik & 1;
    const int nxt = cur ^ 1;
    const int kn  = (ik + 1) * 32;

    if constexpr (sizeof(AT) == 4) {
      // convert + store this step's A registers into As[cur]
#pragma unroll
      for (int i = 0; i < 4; ++i) {
        bf16x4 hv;
        hv[0] = (__bf16)areg[i][0]; hv[1] = (__bf16)areg[i][1];
        hv[2] = (__bf16)areg[i][2]; hv[3] = (__bf16)areg[i][3];
        *(bf16x4*)(As + cur * TILE + (ar + 32 * i) * STR + ac4) = hv;
      }
      // prefetch next step's A into registers
      if (ik + 1 < nk) {
#pragma unroll
        for (int i = 0; i < 4; ++i)
          areg[i] = *(const f32x4*)((const float*)A +
                                    (size_t)(m0 + ar + 32 * i) * K + kn + ac4);
      }
    }
    // issue async fills for the next tile, then fence the current one
    if (ik + 1 < nk) {
      if constexpr (sizeof(AT) == 2) {
        const __bf16* srcA = (const __bf16*)A + (size_t)(m0 + br) * K + kn + bq;
        g2lds16(srcA,     As + nxt * TILE + br * STR + bq);
        g2lds16(srcA + 8, As + nxt * TILE + br * STR + bq + 8);
      }
      const __bf16* srcB = Bt + (size_t)(n0 + br) * K + kn + bq;
      g2lds16(srcB,     Bs + nxt * TILE + br * STR + bq);
      g2lds16(srcB + 8, Bs + nxt * TILE + br * STR + bq + 8);
      if constexpr (sizeof(AT) == 2) WAIT_ASYNC(4); else WAIT_ASYNC(2);
    } else {
      WAIT_ASYNC(0);
    }
    __syncthreads();

    const __bf16* Ab = As + cur * TILE;
    const __bf16* Bb = Bs + cur * TILE;
    v16bf af[2], bfr[4];
#pragma unroll
    for (int mi = 0; mi < 2; ++mi)
      af[mi] = ldA(Ab + (wm * 32 + mi * 16 + hl) * STR + hi * 8);
#pragma unroll
    for (int ni = 0; ni < 4; ++ni)
      bfr[ni] = ldB(Bb + (wn * 64 + ni * 16 + hl) * STR + hi * 8);
#pragma unroll
    for (int mi = 0; mi < 2; ++mi)
#pragma unroll
      for (int ni = 0; ni < 4; ++ni)
        acc[mi][ni] = wmma_bf16(af[mi], bfr[ni], acc[mi][ni]);
    __syncthreads();
  }

  // ---- epilogue: bias add + store ----
  float bcol[4];
#pragma unroll
  for (int ni = 0; ni < 4; ++ni) bcol[ni] = bias[n0 + wn * 64 + ni * 16 + hl];
#pragma unroll
  for (int mi = 0; mi < 2; ++mi)
#pragma unroll
    for (int ni = 0; ni < 4; ++ni) {
      const int col = n0 + wn * 64 + ni * 16 + hl;
#pragma unroll
      for (int r = 0; r < 8; ++r) {
        const int row  = m0 + wm * 32 + mi * 16 + hi * 8 + r;
        const float vl = acc[mi][ni][r] + bcol[ni];
        if constexpr (sizeof(OT) == 2)
          Out[(size_t)row * N + col] = (OT)vl;
        else
          Out[(size_t)row * N + col] = vl;
      }
    }
}

// ---------------------------------------------------------------------------
// Flash attention:  per block: one (b,h), 128 L-rows; stream S in 128 chunks.
// 8 waves x 16 rows; online softmax; all matmuls via v_wmma_f32_16x16x32_bf16.
// K-chunks are async double-buffered (global_load_async_to_lds_b128).
// ---------------------------------------------------------------------------
__global__ __launch_bounds__(256, 1)
void attn_kernel(const __bf16* __restrict__ Q, const __bf16* __restrict__ Kb,
                 const __bf16* __restrict__ Vb, __bf16* __restrict__ O) {
  constexpr int KSTR  = 112;              // halfs; 224B rows (16B mult)
  constexpr int VSTR  = 136;              // halfs; 272B rows (16B mult)
  constexpr int PSTR  = 136;
  constexpr int KTILE = 128 * KSTR;
  __shared__ __bf16 Ks[2 * KTILE];        // K chunk  [s][e]   (double buffered)
  __shared__ __bf16 Vt[EH * VSTR];        // V chunk  [e][s]   (transposed)
  __shared__ __bf16 Ps[128 * PSTR];       // P        [l][s]

  const int tid  = threadIdx.x;
  const int lane = tid & 31;
  const int w    = tid >> 5;              // 8 waves, 16 L-rows each
  const int hl   = lane & 15;
  const int hi   = lane >> 4;
  const int bh   = blockIdx.y;
  const int b    = bh >> 3;               // / NH
  const int h    = bh & 7;
  const int l0   = blockIdx.x * 128;

  const int sfs   = tid >> 1;             // staging row 0..127
  const int sfp   = (tid & 1) * 48;       // staging half-row (48 halfs)

  // Preload this wave's Q fragments (16 x 96 -> 3 K-chunks), kept in VGPRs.
  const __bf16* qrow = Q + (size_t)(b * Lseq + l0 + w * 16 + hl) * DM + h * EH;
  v16bf qa[3];
#pragma unroll
  for (int kc = 0; kc < 3; ++kc) qa[kc] = ldA(qrow + kc * 32 + hi * 8);

  float m_run[8], l_run[8];
  v8f o_acc[6];
#pragma unroll
  for (int r = 0; r < 8; ++r) { m_run[r] = -3.0e38f; l_run[r] = 0.f; }
#pragma unroll
  for (int ne = 0; ne < 6; ++ne) o_acc[ne] = {};

  const float SC = 0.10206207f * 1.44269504f;   // (1/sqrt(E)) * log2(e)

  // ---- prologue: async-stage K chunk 0 ----
  {
    const __bf16* src = Kb + (size_t)sfs * DM + h * EH + sfp;
    __bf16* dst = Ks + sfs * KSTR + sfp;
#pragma unroll
    for (int i = 0; i < 6; ++i) g2lds16(src + i * 8, dst + i * 8);
  }

  for (int ch = 0; ch < Sseq / 128; ++ch) {
    const int s0  = ch * 128;
    const int cur = ch & 1;

    // ---- stage V chunk transposed [e][s] (manual; needs transpose) ----
    {
      const __bf16* src = Vb + (size_t)(s0 + sfs) * DM + h * EH + sfp;
#pragma unroll
      for (int i = 0; i < 6; ++i) {
        bf16x8 v = *(const bf16x8*)(src + i * 8);
#pragma unroll
        for (int j = 0; j < 8; ++j) Vt[(sfp + i * 8 + j) * VSTR + sfs] = v[j];
      }
    }
    // ---- async-stage the NEXT K chunk while this one is consumed ----
    if (ch + 1 < Sseq / 128) {
      const __bf16* src = Kb + (size_t)(s0 + 128 + sfs) * DM + h * EH + sfp;
      __bf16* dst = Ks + (cur ^ 1) * KTILE + sfs * KSTR + sfp;
#pragma unroll
      for (int i = 0; i < 6; ++i) g2lds16(src + i * 8, dst + i * 8);
      WAIT_ASYNC(6);                      // current chunk's 6 ops complete
    } else {
      WAIT_ASYNC(0);
    }
    __syncthreads();

    // ---- scores: 16 x 128 per wave = 24 WMMAs ----
    const __bf16* Kcur = Ks + cur * KTILE;
    v8f sc[8];
#pragma unroll
    for (int ni = 0; ni < 8; ++ni) sc[ni] = {};
#pragma unroll
    for (int kc = 0; kc < 3; ++kc)
#pragma unroll
      for (int ni = 0; ni < 8; ++ni)
        sc[ni] = wmma_bf16(qa[kc],
                           ldB(Kcur + (ni * 16 + hl) * KSTR + kc * 32 + hi * 8),
                           sc[ni]);

    // ---- online softmax (row stats are wave-private; 16-lane reductions) ----
    float alpha[8], rsum[8];
#pragma unroll
    for (int r = 0; r < 8; ++r) {
      float cm = sc[0][r];
#pragma unroll
      for (int ni = 1; ni < 8; ++ni) cm = fmaxf(cm, sc[ni][r]);
      cm = fmaxf(cm, __shfl_xor(cm, 1, 32));
      cm = fmaxf(cm, __shfl_xor(cm, 2, 32));
      cm = fmaxf(cm, __shfl_xor(cm, 4, 32));
      cm = fmaxf(cm, __shfl_xor(cm, 8, 32));
      const float mn = fmaxf(m_run[r], cm);
      alpha[r] = exp2f((m_run[r] - mn) * SC);
      m_run[r] = mn;
      rsum[r]  = 0.f;
    }
    const int prow = w * 16 + hi * 8;
#pragma unroll
    for (int ni = 0; ni < 8; ++ni)
#pragma unroll
      for (int r = 0; r < 8; ++r) {
        const float p = exp2f((sc[ni][r] - m_run[r]) * SC);
        rsum[r] += p;
        Ps[(prow + r) * PSTR + ni * 16 + hl] = (__bf16)p;
      }
#pragma unroll
    for (int r = 0; r < 8; ++r) {
      float rs = rsum[r];
      rs += __shfl_xor(rs, 1, 32);
      rs += __shfl_xor(rs, 2, 32);
      rs += __shfl_xor(rs, 4, 32);
      rs += __shfl_xor(rs, 8, 32);
      l_run[r] = l_run[r] * alpha[r] + rs;
    }
#pragma unroll
    for (int ne = 0; ne < 6; ++ne)
#pragma unroll
      for (int r = 0; r < 8; ++r) o_acc[ne][r] *= alpha[r];

    // ---- O += P*V : K-dim = s (4 chunks of 32) = 24 WMMAs ----
#pragma unroll
    for (int kc = 0; kc < 4; ++kc) {
      const v16bf pa = ldA(Ps + (w * 16 + hl) * PSTR + kc * 32 + hi * 8);
#pragma unroll
      for (int ne = 0; ne < 6; ++ne)
        o_acc[ne] = wmma_bf16(pa,
                              ldB(Vt + (ne * 16 + hl) * VSTR + kc * 32 + hi * 8),
                              o_acc[ne]);
    }
    __syncthreads();
  }

  // ---- normalize and store rep (bf16) ----
#pragma unroll
  for (int r = 0; r < 8; ++r) {
    const float inv  = 1.0f / l_run[r];
    const size_t row = (size_t)(b * Lseq + l0 + w * 16 + hi * 8 + r);
#pragma unroll
    for (int ne = 0; ne < 6; ++ne)
      O[row * DM + h * EH + ne * 16 + hl] = (__bf16)(o_acc[ne][r] * inv);
  }
}

// ---------------------------------------------------------------------------
// Host launcher
// ---------------------------------------------------------------------------
extern "C" void kernel_launch(void* const* d_in, const int* in_sizes, int n_in,
                              void* d_out, int out_size, void* d_ws, size_t ws_size,
                              hipStream_t stream) {
  const float* target = (const float*)d_in[0];
  const float* source = (const float*)d_in[1];
  const float* Wq = (const float*)d_in[2];
  const float* bq = (const float*)d_in[3];
  const float* Wk = (const float*)d_in[4];
  const float* bk = (const float*)d_in[5];
  const float* Wv = (const float*)d_in[6];
  const float* bv = (const float*)d_in[7];
  const float* Wo = (const float*)d_in[8];
  const float* bo = (const float*)d_in[9];
  float* out = (float*)d_out;
  (void)in_sizes; (void)n_in; (void)out_size; (void)ws_size;

  char* ws = (char*)d_ws;
  size_t off = 0;
  auto take = [&](size_t bytes) -> void* {
    void* p = ws + off;
    off += (bytes + 255) & ~(size_t)255;
    return p;
  };
  __bf16* WqT = (__bf16*)take((size_t)DM * DM * 2);     // [768][768]
  __bf16* WkT = (__bf16*)take((size_t)DM * DLL * 2);    // [768][4096]
  __bf16* WvT = (__bf16*)take((size_t)DM * DLL * 2);
  __bf16* WoT = (__bf16*)take((size_t)DLL * DM * 2);    // [4096][768]
  __bf16* Qbf = (__bf16*)take((size_t)Bsz * Lseq * DM * 2);
  __bf16* Kbf = (__bf16*)take((size_t)Sseq * DM * 2);
  __bf16* Vbf = (__bf16*)take((size_t)Sseq * DM * 2);
  __bf16* Rbf = (__bf16*)take((size_t)Bsz * Lseq * DM * 2);

  const dim3 tb(32, 8);
  wtrans_kernel<<<dim3(24, 24),  tb, 0, stream>>>(Wq, WqT, DM, DM);
  wtrans_kernel<<<dim3(24, 128), tb, 0, stream>>>(Wk, WkT, DLL, DM);
  wtrans_kernel<<<dim3(24, 128), tb, 0, stream>>>(Wv, WvT, DLL, DM);
  wtrans_kernel<<<dim3(128, 24), tb, 0, stream>>>(Wo, WoT, DM, DLL);

  // Q = target @ Wq + bq  (M=8192, N=768, K=768) -> bf16
  gemm_wmma_kernel<float, __bf16><<<dim3(6, 64), 256, 0, stream>>>(
      target, WqT, bq, Qbf, Bsz * Lseq, DM, DM);
  // K = source @ Wk + bk  (M=2048, N=768, K=4096) -> bf16
  gemm_wmma_kernel<float, __bf16><<<dim3(6, 16), 256, 0, stream>>>(
      source, WkT, bk, Kbf, Sseq, DM, DLL);
  // V = source @ Wv + bv
  gemm_wmma_kernel<float, __bf16><<<dim3(6, 16), 256, 0, stream>>>(
      source, WvT, bv, Vbf, Sseq, DM, DLL);

  // rep = softmax(Q K^T / sqrt(E)) V   (flash, per (b,h,128 rows))
  attn_kernel<<<dim3(Lseq / 128, Bsz * NH), 256, 0, stream>>>(Qbf, Kbf, Vbf, Rbf);

  // out = rep @ Wo + bo  (M=8192, N=4096, K=768) -> fp32
  gemm_wmma_kernel<__bf16, float><<<dim3(32, 64), 256, 0, stream>>>(
      Rbf, WoT, bo, out, Bsz * Lseq, DLL, DM);
}
